// FourRegionAttention_66571993088896
// MI455X (gfx1250) — matrix operand order repaired and checked by reading, not compiled, not run on hardware
//
#include <hip/hip_runtime.h>
#include <math.h>

typedef __attribute__((ext_vector_type(16))) _Float16 v16h;
typedef __attribute__((ext_vector_type(8)))  float    v8f;

#define BB 2
#define NN 576
#define HH 24
#define DD 64
#define NHEADS 8

// ---------------------------------------------------------------------------
// f32 -> f16 conversion (weights / activations)
// ---------------------------------------------------------------------------
__global__ void k_f32_to_f16(const float* __restrict__ in, _Float16* __restrict__ out, int n) {
  int i = blockIdx.x * blockDim.x + threadIdx.x;
  if (i < n) out[i] = (_Float16)in[i];
}

// ---------------------------------------------------------------------------
// Row LayerNorm over last dim == 64. One wave32 per row; lanes own 2 channels.
// Optional f32 and f16 outputs.
// ---------------------------------------------------------------------------
__global__ void k_ln_rows(const float* __restrict__ in, const float* __restrict__ g,
                          const float* __restrict__ b, float* __restrict__ outf,
                          _Float16* __restrict__ outh, int M) {
  int wid  = (blockIdx.x * blockDim.x + threadIdx.x) >> 5;
  int lane = threadIdx.x & 31;
  if (wid >= M) return;
  const float* row = in + (size_t)wid * 64;
  float x0 = row[lane], x1 = row[lane + 32];
  float s = x0 + x1;
  #pragma unroll
  for (int o = 16; o > 0; o >>= 1) s += __shfl_xor(s, o, 32);
  float m = s * (1.f / 64.f);
  float d0 = x0 - m, d1 = x1 - m;
  float v = d0 * d0 + d1 * d1;
  #pragma unroll
  for (int o = 16; o > 0; o >>= 1) v += __shfl_xor(v, o, 32);
  float r = rsqrtf(v * (1.f / 64.f) + 1e-5f);
  float y0 = d0 * r * g[lane]      + b[lane];
  float y1 = d1 * r * g[lane + 32] + b[lane + 32];
  size_t base = (size_t)wid * 64;
  if (outf) { outf[base + lane] = y0; outf[base + lane + 32] = y1; }
  if (outh) { outh[base + lane] = (_Float16)y0; outh[base + lane + 32] = (_Float16)y1; }
}

// ---------------------------------------------------------------------------
// WMMA GEMM: out[M,N] = act(A[M,K] @ W[N,K]^T + bias) + resid
// A, W in f16 (row-major, ld == K). One 16x16 tile per wave32, K stepped by 32.
// Fragment layouts per CDNA5 ISA 7.12.2:
//   A: M = lane%16; K-halves at klo=(lane<16?0:8): v0..3 -> K=klo+2v{,+1},
//      v4..7 -> K=16+klo+2(v-4){,+1}
//   B: N = lane%16; same K per element as A (kbv = ka)
//   C/D: N = lane%16; M = vgpr + (lane<16?0:8)
// act: 0 = none, 1 = exact GELU. bias/resid/outf/outh nullable.
// ---------------------------------------------------------------------------
__global__ void k_gemm_wmma(const _Float16* __restrict__ A, const _Float16* __restrict__ W,
                            const float* __restrict__ bias, const float* __restrict__ resid,
                            float* __restrict__ outf, _Float16* __restrict__ outh,
                            int M, int N, int K, int act) {
  int ntiles = N >> 4;
  int tiles  = (M >> 4) * ntiles;
  int widx   = blockIdx.x * (blockDim.x >> 5) + (threadIdx.x >> 5);
  if (widx >= tiles) return;
  int tm = widx / ntiles, tn = widx % ntiles;
  int lane = threadIdx.x & 31;
  int hi   = lane >> 4;     // 0 | 1
  int l16  = lane & 15;
  int klo  = hi ? 8 : 0;
  int kb0  = hi ? 16 : 0;

  const _Float16* Arow = A + ((size_t)(tm * 16 + l16)) * K;
  const _Float16* Wrow = W + ((size_t)(tn * 16 + l16)) * K;

  v8f acc = {};
  for (int kc = 0; kc < K; kc += 32) {
    v16h a, b;
    #pragma unroll
    for (int v = 0; v < 8; ++v) {
      int ka = kc + ((v < 4) ? (klo + 2 * v) : (16 + klo + 2 * (v - 4)));
      a[2 * v]     = Arow[ka];
      a[2 * v + 1] = Arow[ka + 1];
      int kbv = ka;
      b[2 * v]     = Wrow[kbv];
      b[2 * v + 1] = Wrow[kbv + 1];
    }
    acc = __builtin_amdgcn_wmma_f32_16x16x32_f16(false, a, false, b, (short)0, acc,
                                                 false, false);
  }

  int gn = tn * 16 + l16;
  float bv = bias ? bias[gn] : 0.f;
  #pragma unroll
  for (int r = 0; r < 8; ++r) {
    int gm = tm * 16 + r + (hi ? 8 : 0);
    float val = acc[r] + bv;
    if (act == 1) val = 0.5f * val * (1.f + erff(val * 0.70710678118654752440f));
    size_t oidx = (size_t)gm * N + gn;
    if (resid) val += resid[oidx];
    if (outf) outf[oidx] = val;
    if (outh) outh[oidx] = (_Float16)val;
  }
}

// ---------------------------------------------------------------------------
// prep: c0 = cls*(1+pw)+pb ; yc = LN(c0,a_ln) ; qkv_c = yc @ a_Wqkv^T ;
// s_cc[h] ; biasE = a_bo + c0 ; effg/effb = pe_ln2 folded with pos affine.
// Single block.
// ---------------------------------------------------------------------------
__global__ void k_prep(const float* cls_tok, const float* pos_w, const float* pos_b,
                       const float* a_ln_g, const float* a_ln_b, const float* a_Wqkv,
                       const float* a_bo, const float* pe2g, const float* pe2b,
                       float* c0, float* qkvc, float* scc, float* biasE,
                       float* effg, float* effb) {
  __shared__ float sc0[64], syc[64], sstat[2];
  int t = threadIdx.x;
  if (t < 64) {
    float c = cls_tok[t] * (1.f + pos_w[t]) + pos_b[t];
    sc0[t] = c;
    c0[t] = c;
    biasE[t] = a_bo[t] + c;
    effg[t] = pe2g[t] * (1.f + pos_w[t]);
    effb[t] = pe2b[t] * (1.f + pos_w[t]) + pos_b[t];
  }
  __syncthreads();
  if (t == 0) {
    float m = 0.f;
    for (int i = 0; i < 64; ++i) m += sc0[i];
    m *= (1.f / 64.f);
    float v = 0.f;
    for (int i = 0; i < 64; ++i) { float d = sc0[i] - m; v += d * d; }
    sstat[0] = m;
    sstat[1] = rsqrtf(v * (1.f / 64.f) + 1e-5f);
  }
  __syncthreads();
  if (t < 64) syc[t] = (sc0[t] - sstat[0]) * sstat[1] * a_ln_g[t] + a_ln_b[t];
  __syncthreads();
  for (int o = t; o < 1536; o += blockDim.x) {
    const float* wr = a_Wqkv + (size_t)o * 64;
    float s = 0.f;
    for (int d = 0; d < 64; ++d) s += syc[d] * wr[d];
    qkvc[o] = s;
  }
  __threadfence_block();
  __syncthreads();
  if (t < NHEADS) {
    float s = 0.f;
    for (int d = 0; d < 64; ++d) s += qkvc[t * 64 + d] * qkvc[512 + t * 64 + d];
    scc[t] = s * 0.125f;
  }
}

// ---------------------------------------------------------------------------
// scores: s_tok[bh,n] = 0.125 * q_c[h] . k_t[b,n,h]; M[bh]=max(s,s_cc); pc=exp
// kv layout: [b*n, 1024] with k at col h*64, v at col 512+h*64. 16 blocks.
// ---------------------------------------------------------------------------
__global__ void k_scores(const float* __restrict__ kv, const float* __restrict__ qkvc,
                         const float* __restrict__ scc, float* __restrict__ stok,
                         float* __restrict__ Mmax, float* __restrict__ pcv) {
  int bh = blockIdx.x, b = bh >> 3, h = bh & 7;
  int wave = threadIdx.x >> 5, lane = threadIdx.x & 31;
  __shared__ float red[8];
  float localmax = -1e30f;
  const float* qc = qkvc + h * 64;
  for (int n = wave; n < NN; n += 8) {
    const float* kr = kv + ((size_t)(b * NN + n)) * 1024 + h * 64;
    float s = qc[lane] * kr[lane] + qc[lane + 32] * kr[lane + 32];
    #pragma unroll
    for (int o = 16; o > 0; o >>= 1) s += __shfl_xor(s, o, 32);
    s *= 0.125f;
    if (lane == 0) { stok[(size_t)bh * NN + n] = s; localmax = fmaxf(localmax, s); }
  }
  if (lane == 0) red[wave] = localmax;
  __syncthreads();
  if (threadIdx.x == 0) {
    float m = scc[h];
    for (int w = 0; w < 8; ++w) m = fmaxf(m, red[w]);
    Mmax[bh] = m;
    pcv[bh] = expf(scc[h] - m);
  }
}

// ---------------------------------------------------------------------------
// Build table T[bh, i*24+j, c] : c<64 -> p*v_t ; c==64 -> p  (p = exp(s - M))
// ---------------------------------------------------------------------------
__global__ void k_build_T(const float* __restrict__ kv, const float* __restrict__ stok,
                          const float* __restrict__ Mmax, float* __restrict__ T) {
  int idx = blockIdx.x * blockDim.x + threadIdx.x;
  const int total = 16 * NN * 65;
  if (idx >= total) return;
  int c = idx % 65, rest = idx / 65;
  int n = rest % NN, bh = rest / NN;
  int b = bh >> 3, h = bh & 7;
  float p = expf(stok[(size_t)bh * NN + n] - Mmax[bh]);
  float val = (c == 64) ? p : p * kv[((size_t)(b * NN + n)) * 1024 + 512 + h * 64 + c];
  T[idx] = val;
}

// ---------------------------------------------------------------------------
// 2D inclusive prefix sums over the 24x24 grid, 65 channels, per (b,h).
// One block per bh; phase1 row scans (j), phase2 column scans (i).
// ---------------------------------------------------------------------------
__global__ void k_prefix(float* __restrict__ T) {
  float* base = T + (size_t)blockIdx.x * NN * 65;
  for (int t = threadIdx.x; t < 65 * HH; t += blockDim.x) {
    int c = t % 65, i = t / 65;
    float s = 0.f;
    for (int j = 0; j < HH; ++j) {
      size_t off = ((size_t)(i * HH + j)) * 65 + c;
      s += base[off];
      base[off] = s;
    }
  }
  __threadfence_block();
  __syncthreads();
  for (int t = threadIdx.x; t < 65 * HH; t += blockDim.x) {
    int c = t % 65, j = t / 65;
    float s = 0.f;
    for (int i = 0; i < HH; ++i) {
      size_t off = ((size_t)(i * HH + j)) * 65 + c;
      s += base[off];
      base[off] = s;
    }
  }
}

// ---------------------------------------------------------------------------
// Region combine: o[b,l,r, h*64+d] = (regSum(d) + pc*v_c[d]) / (regSum(64)+pc)
// Regions via inclusion-exclusion on the prefix table. Output directly f16.
// ---------------------------------------------------------------------------
__global__ void k_regions(const float* __restrict__ T, const float* __restrict__ qkvc,
                          const float* __restrict__ pcv, _Float16* __restrict__ o_h) {
  int idx = blockIdx.x * blockDim.x + threadIdx.x;
  const int total = BB * NN * 4 * 512;
  if (idx >= total) return;
  int col = idx & 511;   int d = col & 63;   int h = col >> 6;
  int rest = idx >> 9;   int r = rest & 3;   rest >>= 2;
  int l = rest % NN;     int b = rest / NN;
  int i = l / HH, j = l % HH;
  int bh = b * 8 + h;
  const float* base = T + (size_t)bh * NN * 65;
  auto S = [&](int ii, int jj, int c) -> float {
    if (ii < 0 || jj < 0) return 0.f;
    return base[((size_t)(ii * HH + jj)) * 65 + c];
  };
  auto reg = [&](int c) -> float {
    switch (r) {
      case 0:  return S(i, j, c);
      case 1:  return S(i, HH - 1, c) - S(i, j - 1, c);
      case 2:  return S(HH - 1, j, c) - S(i - 1, j, c);
      default: return S(HH - 1, HH - 1, c) - S(i - 1, HH - 1, c)
                    - S(HH - 1, j - 1, c) + S(i - 1, j - 1, c);
    }
  };
  float pc  = pcv[bh];
  float num = reg(d)  + pc * qkvc[1024 + h * 64 + d];   // + pc * v_c
  float den = reg(64) + pc;
  o_h[idx] = (_Float16)(num / den);
}

// ---------------------------------------------------------------------------
// Region aggregator: per row (b,l,h): 4 scores, softmax, weighted sum of v2.
// One wave32 per row. Rows ordered (b,l,h) -> output is [b,n,512] contiguous.
// ---------------------------------------------------------------------------
__global__ void k_agg(const float* __restrict__ q2, const float* __restrict__ k2,
                      const float* __restrict__ v2, _Float16* __restrict__ agg_h) {
  int wid  = (blockIdx.x * blockDim.x + threadIdx.x) >> 5;
  int lane = threadIdx.x & 31;
  if (wid >= BB * NN * NHEADS) return;
  int bl = wid >> 3;                 // b*576 + l
  const float* qr = q2 + (size_t)wid * 64;
  float q0 = qr[lane], q1 = qr[lane + 32];
  float sc[4];
  #pragma unroll
  for (int r = 0; r < 4; ++r) {
    const float* kr = k2 + ((size_t)(bl * 4 + r)) * 64;
    float s = q0 * kr[lane] + q1 * kr[lane + 32];
    #pragma unroll
    for (int o = 16; o > 0; o >>= 1) s += __shfl_xor(s, o, 32);
    sc[r] = s * 0.35355339059327373f;   // (DIM/HEADS)^-0.5 = 8^-0.5
  }
  float mx = fmaxf(fmaxf(sc[0], sc[1]), fmaxf(sc[2], sc[3]));
  float e0 = expf(sc[0] - mx), e1 = expf(sc[1] - mx);
  float e2 = expf(sc[2] - mx), e3 = expf(sc[3] - mx);
  float inv = 1.f / (e0 + e1 + e2 + e3);
  float a0 = e0 * inv, a1 = e1 * inv, a2 = e2 * inv, a3 = e3 * inv;
  const float* vb = v2 + (size_t)(bl * 4) * 64;
  float o0 = a0 * vb[lane]      + a1 * vb[64 + lane]
           + a2 * vb[128 + lane] + a3 * vb[192 + lane];
  float o1 = a0 * vb[lane + 32]      + a1 * vb[64 + lane + 32]
           + a2 * vb[128 + lane + 32] + a3 * vb[192 + lane + 32];
  agg_h[(size_t)wid * 64 + lane]      = (_Float16)o0;
  agg_h[(size_t)wid * 64 + lane + 32] = (_Float16)o1;
}

// ---------------------------------------------------------------------------
extern "C" void kernel_launch(void* const* d_in, const int* in_sizes, int n_in,
                              void* d_out, int out_size, void* d_ws, size_t ws_size,
                              hipStream_t stream) {
  (void)in_sizes; (void)n_in; (void)out_size; (void)ws_size;
  const float* x       = (const float*)d_in[0];
  const float* ln_g    = (const float*)d_in[1];
  const float* ln_b    = (const float*)d_in[2];
  const float* Wqkv    = (const float*)d_in[3];
  const float* bqkv    = (const float*)d_in[4];
  const float* Wout    = (const float*)d_in[5];
  const float* bout    = (const float*)d_in[6];
  const float* pe_ln1g = (const float*)d_in[7];
  const float* pe_ln1b = (const float*)d_in[8];
  const float* pe_W    = (const float*)d_in[9];
  const float* pe_b    = (const float*)d_in[10];
  const float* pe_ln2g = (const float*)d_in[11];
  const float* pe_ln2b = (const float*)d_in[12];
  const float* cls_tok = (const float*)d_in[13];
  const float* pos_w   = (const float*)d_in[14];
  const float* pos_b   = (const float*)d_in[15];
  const float* a_ln_g  = (const float*)d_in[16];
  const float* a_ln_b  = (const float*)d_in[17];
  const float* a_Wqkv  = (const float*)d_in[18];
  const float* a_Wo    = (const float*)d_in[19];
  const float* a_bo    = (const float*)d_in[20];
  const float* f_ln_g  = (const float*)d_in[21];
  const float* f_ln_b  = (const float*)d_in[22];
  const float* f_W1    = (const float*)d_in[23];
  const float* f_b1    = (const float*)d_in[24];
  const float* f_W2    = (const float*)d_in[25];
  const float* f_b2    = (const float*)d_in[26];
  const float* t_ln_g  = (const float*)d_in[27];
  const float* t_ln_b  = (const float*)d_in[28];
  const float* head_W  = (const float*)d_in[29];
  const float* head_b  = (const float*)d_in[30];
  const float* ra_Wq   = (const float*)d_in[31];
  const float* ra_Wk   = (const float*)d_in[32];
  const float* ra_Wv   = (const float*)d_in[33];
  const float* ra_Wo   = (const float*)d_in[34];
  const float* ra_bo   = (const float*)d_in[35];
  float* out = (float*)d_out;

  const int M1 = BB * NN;       // 1152 token rows
  const int M4 = BB * NN * 4;   // 4608 region rows
  const int M8 = BB * NN * 8;   // 9216 (b,l,h) rows

  // deterministic bump allocator over d_ws
  size_t off = 0;
  auto alloc = [&](size_t bytes) -> void* {
    void* p = (char*)d_ws + off;
    off += (bytes + 255) & ~(size_t)255;
    return p;
  };
  // f16 weights
  _Float16* wq_h    = (_Float16*)alloc(512 * 64 * 2);
  _Float16* peW_h   = (_Float16*)alloc(64 * 64 * 2);
  _Float16* aWkv_h  = (_Float16*)alloc(1024 * 64 * 2);
  _Float16* aWo_h   = (_Float16*)alloc(64 * 512 * 2);
  _Float16* fW1_h   = (_Float16*)alloc(64 * 64 * 2);
  _Float16* fW2_h   = (_Float16*)alloc(64 * 64 * 2);
  _Float16* headW_h = (_Float16*)alloc(64 * 64 * 2);
  _Float16* raWq_h  = (_Float16*)alloc(64 * 64 * 2);
  _Float16* raWk_h  = (_Float16*)alloc(64 * 64 * 2);
  _Float16* raWv_h  = (_Float16*)alloc(64 * 64 * 2);
  _Float16* raWo_h  = (_Float16*)alloc(64 * 64 * 2);
  _Float16* Wout_h  = (_Float16*)alloc(64 * 512 * 2);
  // small prep buffers
  float* c0    = (float*)alloc(64 * 4);
  float* qkvc  = (float*)alloc(1536 * 4);
  float* scc   = (float*)alloc(16 * 4);
  float* biasE = (float*)alloc(64 * 4);
  float* effg  = (float*)alloc(64 * 4);
  float* effb  = (float*)alloc(64 * 4);
  // activations
  float*     x1_f   = (float*)alloc((size_t)M1 * 64 * 4);
  _Float16*  x1_h   = (_Float16*)alloc((size_t)M1 * 64 * 2);
  _Float16*  x2_h   = (_Float16*)alloc((size_t)M1 * 64 * 2);
  float*     tpe_f  = (float*)alloc((size_t)M1 * 64 * 4);
  float*     e_f    = (float*)alloc((size_t)M1 * 64 * 4);
  _Float16*  yt_h   = (_Float16*)alloc((size_t)M1 * 64 * 2);
  float*     kv_f   = (float*)alloc((size_t)M1 * 1024 * 4);
  _Float16*  q_h    = (_Float16*)alloc((size_t)M1 * 512 * 2);
  float*     stok   = (float*)alloc((size_t)16 * NN * 4);
  float*     Mmax   = (float*)alloc(16 * 4);
  float*     pcv    = (float*)alloc(16 * 4);
  float*     Ttab   = (float*)alloc((size_t)16 * NN * 65 * 4);
  _Float16*  o_h    = (_Float16*)alloc((size_t)M4 * 512 * 2);
  float*     cls1   = (float*)alloc((size_t)M4 * 64 * 4);
  _Float16*  c1ln_h = (_Float16*)alloc((size_t)M4 * 64 * 2);
  _Float16*  ffh_h  = (_Float16*)alloc((size_t)M4 * 64 * 2);
  float*     cls2   = (float*)alloc((size_t)M4 * 64 * 4);
  _Float16*  c2ln_h = (_Float16*)alloc((size_t)M4 * 64 * 2);
  _Float16*  reg_h  = (_Float16*)alloc((size_t)M4 * 64 * 2);
  float*     k2_f   = (float*)alloc((size_t)M4 * 64 * 4);
  float*     v2_f   = (float*)alloc((size_t)M4 * 64 * 4);
  float*     q2_f   = (float*)alloc((size_t)M8 * 64 * 4);
  _Float16*  agg_h  = (_Float16*)alloc((size_t)M8 * 64 * 2);
  _Float16*  outp_h = (_Float16*)alloc((size_t)M8 * 64 * 2);

  auto conv = [&](const float* src, _Float16* dst, int n) {
    k_f32_to_f16<<<(n + 255) / 256, 256, 0, stream>>>(src, dst, n);
  };
  auto ln = [&](const float* in, const float* g, const float* b, float* of,
                _Float16* oh, int M) {
    k_ln_rows<<<(M * 32 + 255) / 256, 256, 0, stream>>>(in, g, b, of, oh, M);
  };
  auto gemm = [&](const _Float16* A, const _Float16* W, const float* bias,
                  const float* resid, float* of, _Float16* oh, int M, int N, int K,
                  int act) {
    int tiles = (M / 16) * (N / 16);
    k_gemm_wmma<<<(tiles + 7) / 8, 256, 0, stream>>>(A, W, bias, resid, of, oh, M, N, K,
                                                     act);
  };

  // --- weight conversion ---
  conv(Wqkv, wq_h, 512 * 64);                 // first 512 rows of Wqkv (q only)
  conv(pe_W, peW_h, 64 * 64);
  conv(a_Wqkv + 512 * 64, aWkv_h, 1024 * 64); // k,v rows of a_Wqkv
  conv(a_Wo, aWo_h, 64 * 512);
  conv(f_W1, fW1_h, 64 * 64);
  conv(f_W2, fW2_h, 64 * 64);
  conv(head_W, headW_h, 64 * 64);
  conv(ra_Wq, raWq_h, 64 * 64);
  conv(ra_Wk, raWk_h, 64 * 64);
  conv(ra_Wv, raWv_h, 64 * 64);
  conv(ra_Wo, raWo_h, 64 * 64);
  conv(Wout, Wout_h, 64 * 512);

  // --- prep (cls path, fused biases, pe_ln2+pos fold) ---
  k_prep<<<1, 256, 0, stream>>>(cls_tok, pos_w, pos_b, a_ln_g, a_ln_b, a_Wqkv, a_bo,
                                pe_ln2g, pe_ln2b, c0, qkvc, scc, biasE, effg, effb);

  // --- token path ---
  ln(x, ln_g, ln_b, x1_f, x1_h, M1);                 // x1 = LN(x)
  ln(x1_f, pe_ln1g, pe_ln1b, nullptr, x2_h, M1);     // LN(x1, pe_ln1)
  gemm(x2_h, peW_h, pe_b, nullptr, tpe_f, nullptr, M1, 64, 64, 0);
  ln(tpe_f, effg, effb, e_f, nullptr, M1);           // e = pos-affine(LN(.,pe_ln2))
  ln(e_f, a_ln_g, a_ln_b, nullptr, yt_h, M1);        // yt = LN(e, a_ln)
  gemm(yt_h, aWkv_h, nullptr, nullptr, kv_f, nullptr, M1, 1024, 64, 0);  // k_t|v_t
  gemm(x1_h, wq_h, bqkv, nullptr, nullptr, q_h, M1, 512, 64, 0);         // q

  // --- DWA via 2D prefix sums (softmax shift-invariance) ---
  k_scores<<<16, 256, 0, stream>>>(kv_f, qkvc, scc, stok, Mmax, pcv);
  {
    int tot = 16 * NN * 65;
    k_build_T<<<(tot + 255) / 256, 256, 0, stream>>>(kv_f, stok, Mmax, Ttab);
  }
  k_prefix<<<16, 256, 0, stream>>>(Ttab);
  {
    int tot = BB * NN * 4 * 512;
    k_regions<<<(tot + 255) / 256, 256, 0, stream>>>(Ttab, qkvc, pcv, o_h);
  }

  // --- region transformer ---
  gemm(o_h, aWo_h, biasE, nullptr, cls1, nullptr, M4, 64, 512, 0);   // cls1 (+a_bo+c0)
  ln(cls1, f_ln_g, f_ln_b, nullptr, c1ln_h, M4);
  gemm(c1ln_h, fW1_h, f_b1, nullptr, nullptr, ffh_h, M4, 64, 64, 1); // GELU
  gemm(ffh_h, fW2_h, f_b2, cls1, cls2, nullptr, M4, 64, 64, 0);      // + residual
  ln(cls2, t_ln_g, t_ln_b, nullptr, c2ln_h, M4);
  gemm(c2ln_h, headW_h, head_b, nullptr, nullptr, reg_h, M4, 64, 64, 0); // region
  gemm(reg_h, raWk_h, nullptr, nullptr, k2_f, nullptr, M4, 64, 64, 0);
  gemm(reg_h, raWv_h, nullptr, nullptr, v2_f, nullptr, M4, 64, 64, 0);

  // --- aggregator ---
  gemm(q_h, raWq_h, nullptr, nullptr, q2_f, nullptr, M8, 64, 64, 0);   // q2
  k_agg<<<(M8 * 32 + 255) / 256, 256, 0, stream>>>(q2_f, k2_f, v2_f, agg_h);
  gemm(agg_h, raWo_h, ra_bo, nullptr, nullptr, outp_h, M8, 64, 64, 0); // outp [b,n,512]

  // --- final projection ---
  gemm(outp_h, Wout_h, bout, nullptr, out, nullptr, M1, 64, 512, 0);
}
